// group_explore_15315853377830
// MI455X (gfx1250) — matrix operand order repaired and checked by reading, not compiled, not run on hardware
//
#include <hip/hip_runtime.h>
#include <hip/hip_bf16.h>

typedef _Float16 h8  __attribute__((ext_vector_type(8)));
typedef _Float16 v16h __attribute__((ext_vector_type(16)));
typedef float    v8f  __attribute__((ext_vector_type(8)));

#define NB   16
#define CH   64
#define HH   112
#define WW   112
#define ROWS (NB*HH)                 // 1792 workgroups per conv
#define TEN  ((size_t)NB*HH*WW*CH)   // halves per activation tensor
#define WSW_PER_CONV (4*9*2*32*16)   // swizzled weight halves per 3x3 conv (=36864)

__device__ __forceinline__ float qrelu_f(float x) {
    float t = fminf(fmaxf(x, 0.f), 10.f) * (255.f / 10.f);
    t = rintf(t);                       // jnp.round == nearest-even == rintf
    t = fminf(fmaxf(t, 0.f), 255.f);
    return t * (10.f / 255.f);
}
__device__ __forceinline__ float qht_f(float x) {
    float t = fminf(fmaxf(x, -10.f), 10.f) * (127.f / 10.f);
    t = rintf(t);
    t = fminf(fmaxf(t, -127.f), 127.f);
    return t * (10.f / 127.f);
}

__device__ __forceinline__ v16h mkA(const _Float16* ap, int c0) {
    h8 lo = *(const h8*)(ap + c0);
    h8 hi = *(const h8*)(ap + c0 + 16);
    return __builtin_shufflevector(lo, hi, 0,1,2,3,4,5,6,7,8,9,10,11,12,13,14,15);
}

// ---------------------------------------------------------------------------
// NCHW f32  ->  NHWC f16 (one block per (n,h) row; writes fully coalesced)
// ---------------------------------------------------------------------------
__global__ void __launch_bounds__(256) nchw_to_nhwc_h(const float* __restrict__ x,
                                                      _Float16* __restrict__ xh) {
    const int row = blockIdx.x;
    const int n = row / HH, h = row % HH;
    const size_t dstBase = ((size_t)n * HH + h) * WW * CH;
    for (int idx = threadIdx.x; idx < WW * CH; idx += 256) {
        const int w = idx >> 6, c = idx & 63;
        xh[dstBase + (size_t)w * CH + c] =
            (_Float16)x[(((size_t)n * CH + c) * HH + h) * WW + w];
    }
}

// ---------------------------------------------------------------------------
// Binary weight quant + swizzle into per-lane WMMA-B fragment order.
// Layout: wsw[i][nt(4)][kyx(9)][kc(2)][lane(32)][j(16)]
//   lane = khalf*16 + (oc%16);  channel c = kc*32 + (j>>3)*16 + khalf*8 + (j&7)
// ---------------------------------------------------------------------------
__global__ void __launch_bounds__(64) prep_bin(const float* __restrict__ conv_w,
                                               _Float16* __restrict__ wsw) {
    const int oc = blockIdx.x;          // 0..63
    const int i  = blockIdx.y;          // 0..3
    const int tid = threadIdx.x;        // 64 threads
    const float* w = conv_w + ((size_t)i * CH + oc) * (CH * 9);  // [c][ky][kx]
    __shared__ float red[64];
    float s = 0.f;
    for (int e = tid; e < 576; e += 64) s += fabsf(w[e]);
    red[tid] = s;
    __syncthreads();
    for (int st = 32; st > 0; st >>= 1) { if (tid < st) red[tid] += red[tid + st]; __syncthreads(); }
    const float scale = fmaxf(red[0] * (1.f / 576.f), 2e-16f);
    const int nt = oc >> 4, lm = oc & 15;
    _Float16* base = wsw + (size_t)i * WSW_PER_CONV;
    for (int e = tid; e < 576; e += 64) {
        const int kyx = e / 64, c = e % 64;
        const int kc = c >> 5, cr = c & 31, khalf = (cr >> 3) & 1, j = (cr & 7) | ((cr >> 4) << 3);
        const int lane = khalf * 16 + lm;
        const float wv = w[c * 9 + kyx];
        base[((((size_t)nt * 9 + kyx) * 2 + kc) * 32 + lane) * 16 + j] =
            (_Float16)(wv >= 0.f ? scale : -scale);
    }
}

// int8 narrow-range quant of the 1x1 weights, same B-fragment swizzle.
// Layout: dsw[nt(4)][kc(2)][lane(32)][j(16)]
__global__ void __launch_bounds__(64) prep_ds(const float* __restrict__ ds_w,
                                              _Float16* __restrict__ dsw) {
    const int oc = blockIdx.x, tid = threadIdx.x;
    const float* w = ds_w + (size_t)oc * CH;
    __shared__ float red[64];
    red[tid] = fabsf(w[tid]);
    __syncthreads();
    for (int st = 32; st > 0; st >>= 1) { if (tid < st) red[tid] = fmaxf(red[tid], red[tid + st]); __syncthreads(); }
    const float scale = fmaxf(red[0], 2e-16f) * (1.f / 127.f);
    const int c = tid;
    const float q = fminf(fmaxf(rintf(w[c] / scale), -127.f), 127.f) * scale;
    const int kc = c >> 5, cr = c & 31, khalf = (cr >> 3) & 1, j = (cr & 7) | ((cr >> 4) << 3);
    const int nt = oc >> 4, lane = khalf * 16 + (oc & 15);
    dsw[(((size_t)nt * 2 + kc) * 32 + lane) * 16 + j] = (_Float16)q;
}

// ---------------------------------------------------------------------------
// 1x1 conv (identity0): M=pixels, N=64, K=64 -> 2 WMMA per tile.
// nt fixed per wave so both B fragments hoist out of the M loop.
// Epilogue: id0h = qhtanh(bn4(acc))  (stored f16 NHWC)
// ---------------------------------------------------------------------------
__global__ void __launch_bounds__(256) conv1x1_wmma(const _Float16* __restrict__ in,
                                                    const _Float16* __restrict__ dsw,
                                                    const float* __restrict__ g,
                                                    const float* __restrict__ b,
                                                    const float* __restrict__ mu,
                                                    const float* __restrict__ vr,
                                                    _Float16* __restrict__ out) {
    const int row = blockIdx.x;
    const int n = row / HH, h = row % HH;
    const size_t rowBase = ((size_t)n * HH + h) * WW * CH;
    const int wave = threadIdx.x >> 5, lane = threadIdx.x & 31;
    const int lhalf = lane >> 4, lmod = lane & 15;
    const int nt = wave & 3;
    const int mt0 = (wave >> 2) ? 4 : 0;
    const int mtN = (wave >> 2) ? 3 : 4;

    const v16h B0 = *(const v16h*)(dsw + (((size_t)nt * 2 + 0) * 32 + lane) * 16);
    const v16h B1 = *(const v16h*)(dsw + (((size_t)nt * 2 + 1) * 32 + lane) * 16);
    const int ch = nt * 16 + lmod;
    const float inv = g[ch] * rsqrtf(vr[ch] + 1e-5f);
    const float off = b[ch] - mu[ch] * inv;

    for (int mi = 0; mi < mtN; ++mi) {
        const int mt = mt0 + mi;
        const _Float16* ap = in + rowBase + (size_t)(mt * 16 + lmod) * CH;
        v8f acc{};
        acc = __builtin_amdgcn_wmma_f32_16x16x32_f16(false, mkA(ap, lhalf * 8),      false, B0, (short)0, acc, false, false);
        acc = __builtin_amdgcn_wmma_f32_16x16x32_f16(false, mkA(ap, 32 + lhalf * 8), false, B1, (short)0, acc, false, false);
#pragma unroll
        for (int r = 0; r < 8; ++r) {
            const int mm = mt * 16 + r + lhalf * 8;
            out[rowBase + (size_t)mm * CH + ch] = (_Float16)qht_f(acc[r] * inv + off);
        }
    }
}

// ---------------------------------------------------------------------------
// 3x3 binary-weight conv, implicit GEMM, fused BN + quantized skip epilogues.
// One workgroup per (n,h) output row; K = 9*64 = 18 WMMA steps per tile.
// Each wave owns one N-tile and runs two M-tiles concurrently: the shared B
// fragment is loaded once, and the two independent WMMAs fill the F16
// WMMA->WMMA hazard slots (no v_nop stalls).
// Activation rows are staged memory->LDS with global_load_async_to_lds_b128
// (ASYNCcnt), halo borders zero-filled with ds_store.
// MODE 0: s1  = qrelu(qht(bn0(acc)) + qht(x_center))
// MODE 1: a   = qht(bn1(acc)) + id0h;  s2 = qrelu(qht(a) + qht(s1_center))
// MODE 2: s3  = qrelu(qht(bn2(acc)) + qht(s2_center))
// MODE 3: o   = qht(bn3(acc)) + qht(s2);  out = qrelu(qht(s3_center) + qht(o))  [f32 NCHW]
// ---------------------------------------------------------------------------
template <int MODE>
__global__ void __launch_bounds__(256) conv3x3_wmma(const _Float16* __restrict__ in,
                                                    const _Float16* __restrict__ wsw,
                                                    const _Float16* __restrict__ aux,
                                                    const float* __restrict__ g,
                                                    const float* __restrict__ b,
                                                    const float* __restrict__ mu,
                                                    const float* __restrict__ vr,
                                                    _Float16* __restrict__ outH,
                                                    float* __restrict__ outF) {
    const int row = blockIdx.x;
    const int n = row / HH, h = row % HH;
    __shared__ alignas(16) _Float16 sAct[3 * 114 * CH];   // 43.8 KB: 3 rows + W halo

    // Stage 3 halo-padded input rows: async DMA memory->LDS for in-bounds
    // chunks (sAct is the only LDS object -> its base offset is 0), ds_store
    // zero-fill for out-of-image halo chunks.
    for (int idx = threadIdx.x; idx < 3 * 114 * 8; idx += 256) {
        const int r  = idx / (114 * 8);
        const int pr = idx % (114 * 8);
        const int p  = pr >> 3, cc = pr & 7;
        const int hs = h - 1 + r, ws = p - 1;
        const int elem = (r * 114 + p) * CH + cc * 8;
        if (hs >= 0 && hs < HH && ws >= 0 && ws < WW) {
            const unsigned lds = (unsigned)(elem * sizeof(_Float16));
            const _Float16* gp = in + (((size_t)n * HH + hs) * WW + ws) * CH + cc * 8;
            asm volatile("global_load_async_to_lds_b128 %0, %1, off"
                         :: "v"(lds), "v"(gp) : "memory");
        } else {
            h8 z{};
            *(h8*)(sAct + elem) = z;
        }
    }
    asm volatile("s_wait_asynccnt 0x0" ::: "memory");
    __syncthreads();

    const int wave = threadIdx.x >> 5, lane = threadIdx.x & 31;
    const int lhalf = lane >> 4, lmod = lane & 15;
    const int nt = wave & 3;
    if (lane == 0) __builtin_prefetch(wsw + ((size_t)nt * 9 * 2 * 32 * 16), 0, 3);

    const int ch = nt * 16 + lmod;
    const float inv = g[ch] * rsqrtf(vr[ch] + 1e-5f);
    const float off = b[ch] - mu[ch] * inv;
    const size_t rowBase = ((size_t)n * HH + h) * WW * CH;

    // Dual-accumulator K loop: one B load feeds two M-tiles.
    auto runPair = [&](int mt0, int mt1, v8f& a0, v8f& a1) {
#pragma unroll
        for (int kyx = 0; kyx < 9; ++kyx) {
            const int ky = kyx / 3, kx = kyx % 3;
            const _Float16* ap0 = sAct + ((size_t)ky * 114 + (mt0 * 16 + lmod + kx)) * CH;
            const _Float16* ap1 = sAct + ((size_t)ky * 114 + (mt1 * 16 + lmod + kx)) * CH;
#pragma unroll
            for (int kc = 0; kc < 2; ++kc) {
                const int c0 = kc * 32 + lhalf * 8;
                const v16h B = *(const v16h*)(wsw + (((size_t)(nt * 9 + kyx) * 2 + kc) * 32 + lane) * 16);
                const v16h A0 = mkA(ap0, c0);
                const v16h A1 = mkA(ap1, c0);
                a0 = __builtin_amdgcn_wmma_f32_16x16x32_f16(false, A0, false, B, (short)0, a0, false, false);
                a1 = __builtin_amdgcn_wmma_f32_16x16x32_f16(false, A1, false, B, (short)0, a1, false, false);
            }
        }
    };
    auto runSingle = [&](int mt, v8f& a0) {
#pragma unroll
        for (int kyx = 0; kyx < 9; ++kyx) {
            const int ky = kyx / 3, kx = kyx % 3;
            const _Float16* ap = sAct + ((size_t)ky * 114 + (mt * 16 + lmod + kx)) * CH;
#pragma unroll
            for (int kc = 0; kc < 2; ++kc) {
                const int c0 = kc * 32 + lhalf * 8;
                const v16h B = *(const v16h*)(wsw + (((size_t)(nt * 9 + kyx) * 2 + kc) * 32 + lane) * 16);
                a0 = __builtin_amdgcn_wmma_f32_16x16x32_f16(false, mkA(ap, c0), false, B, (short)0, a0, false, false);
            }
        }
    };
    auto epi = [&](int mt, const v8f& acc) {
#pragma unroll
        for (int r = 0; r < 8; ++r) {
            const int m = mt * 16 + r + lhalf * 8;            // output pixel (w coord)
            const float val = acc[r] * inv + off;
            const float center = (float)sAct[((size_t)1 * 114 + (m + 1)) * CH + ch];
            const size_t gofs = rowBase + (size_t)m * CH + ch;
            if constexpr (MODE == 0) {
                outH[gofs] = (_Float16)qrelu_f(qht_f(val) + qht_f(center));
            } else if constexpr (MODE == 1) {
                const float a2 = qht_f(val) + (float)aux[gofs];          // + qht(identity0)
                outH[gofs] = (_Float16)qrelu_f(qht_f(a2) + qht_f(center));
            } else if constexpr (MODE == 2) {
                outH[gofs] = (_Float16)qrelu_f(qht_f(val) + qht_f(center));
            } else {
                const float o  = qht_f(val) + qht_f((float)aux[gofs]);   // out11
                const float gg = qht_f(center) + qht_f(o);               // group_out
                outF[(((size_t)n * CH + ch) * HH + h) * WW + m] = qrelu_f(gg);
            }
        }
    };

    if ((wave >> 2) == 0) {           // M-tiles {0,1,2,3}
        v8f a0{}, a1{};
        runPair(0, 1, a0, a1); epi(0, a0); epi(1, a1);
        v8f b0{}, b1{};
        runPair(2, 3, b0, b1); epi(2, b0); epi(3, b1);
    } else {                          // M-tiles {4,5,6}
        v8f a0{}, a1{};
        runPair(4, 5, a0, a1); epi(4, a0); epi(5, a1);
        v8f c0{};
        runSingle(6, c0); epi(6, c0);
    }
}

// ---------------------------------------------------------------------------
extern "C" void kernel_launch(void* const* d_in, const int* in_sizes, int n_in,
                              void* d_out, int out_size, void* d_ws, size_t ws_size,
                              hipStream_t stream) {
    const float* x      = (const float*)d_in[0];
    const float* conv_w = (const float*)d_in[1];
    const float* ds_w   = (const float*)d_in[2];
    const float* bng    = (const float*)d_in[3];   // (5,64)
    const float* bnb    = (const float*)d_in[4];
    const float* bnm    = (const float*)d_in[5];
    const float* bnv    = (const float*)d_in[6];
    float* out = (float*)d_out;

    _Float16* ws = (_Float16*)d_ws;
    _Float16* xh  = ws;
    _Float16* s1  = xh + TEN;
    _Float16* s2  = s1 + TEN;
    _Float16* s3  = s2 + TEN;
    _Float16* id0 = s3 + TEN;
    _Float16* wsw = id0 + TEN;                 // 4 * 36864 halves
    _Float16* dsw = wsw + 4 * WSW_PER_CONV;    // 4096 halves

    // Weight quant + swizzle; activation layout transform.
    prep_bin<<<dim3(64, 4), 64, 0, stream>>>(conv_w, wsw);
    prep_ds<<<64, 64, 0, stream>>>(ds_w, dsw);
    nchw_to_nhwc_h<<<ROWS, 256, 0, stream>>>(x, xh);

    // identity0 = qht(bn4(conv1x1(x)))
    conv1x1_wmma<<<ROWS, 256, 0, stream>>>(xh, dsw, bng + 4*64, bnb + 4*64,
                                           bnm + 4*64, bnv + 4*64, id0);
    // block0 layer0
    conv3x3_wmma<0><<<ROWS, 256, 0, stream>>>(xh, wsw + 0*WSW_PER_CONV, nullptr,
                                              bng + 0, bnb + 0, bnm + 0, bnv + 0,
                                              s1, nullptr);
    // block0 layer1 (+identity0, +op1=s1)
    conv3x3_wmma<1><<<ROWS, 256, 0, stream>>>(s1, wsw + 1*WSW_PER_CONV, id0,
                                              bng + 1*64, bnb + 1*64, bnm + 1*64, bnv + 1*64,
                                              s2, nullptr);
    // block1 layer0 (+op2=s2)
    conv3x3_wmma<2><<<ROWS, 256, 0, stream>>>(s2, wsw + 2*WSW_PER_CONV, nullptr,
                                              bng + 2*64, bnb + 2*64, bnm + 2*64, bnv + 2*64,
                                              s3, nullptr);
    // block1 layer1 (+identity1=s2, +op3=s3) -> final f32 NCHW
    conv3x3_wmma<3><<<ROWS, 256, 0, stream>>>(s3, wsw + 3*WSW_PER_CONV, s2,
                                              bng + 3*64, bnb + 3*64, bnm + 3*64, bnv + 3*64,
                                              nullptr, out);
    (void)in_sizes; (void)n_in; (void)out_size; (void)ws_size;
}
